// AttentionLayer_45621142618790
// MI455X (gfx1250) — matrix operand order repaired and checked by reading, not compiled, not run on hardware
//
#include <hip/hip_runtime.h>

typedef __attribute__((ext_vector_type(16))) _Float16 v16h;
typedef __attribute__((ext_vector_type(8)))  _Float16 v8h;
typedef __attribute__((ext_vector_type(8)))  float    v8f;

#define DEVINL __device__ __forceinline__

// Problem constants (fixed by the reference).
constexpr int Bb = 4;      // batch
constexpr int Lq = 2048;   // query len
constexpr int Sk = 2048;   // key len
constexpr int Dm = 64;     // model dim
constexpr int Hh = 8;      // heads
constexpr int Hd = 8;      // head dim

DEVINL v8f wmma_f16(v16h a, v16h b, v8f c) {
  // D = A(16x32 f16) * B(32x16 f16) + C(16x16 f32)
  return __builtin_amdgcn_wmma_f32_16x16x32_f16(
      /*neg_a=*/false, a, /*neg_b=*/false, b,
      /*c_mod=*/(short)0, c, /*reuse_a=*/false, /*reuse_b=*/false);
}

// ---------------------------------------------------------------------------
// Kernel 1: fused QKV projection.  y = x @ W.T + b, stored f16 to workspace.
// Block = 128 threads (4 waves); each wave owns a 16-row M tile; WG = 64 rows.
// ---------------------------------------------------------------------------
__global__ void __launch_bounds__(128)
attn_qkv_proj(const float* __restrict__ Xq, const float* __restrict__ Xk,
              const float* __restrict__ Xv,
              const float* __restrict__ Wq, const float* __restrict__ bq,
              const float* __restrict__ Wk, const float* __restrict__ bk,
              const float* __restrict__ Wv, const float* __restrict__ bv,
              _Float16* __restrict__ Q16, _Float16* __restrict__ K16,
              _Float16* __restrict__ V16) {
  __shared__ __align__(32) _Float16 Wl[3][64][64];  // f16 weights, 24 KB

  const int tid = threadIdx.x;
  // Stage all three weight matrices into LDS as f16 (rows of W = columns of B).
  for (int i = tid; i < 3 * 64 * 64; i += 128) {
    const int m = i >> 12, j = i & 4095;
    const float* src = (m == 0) ? Wq : (m == 1) ? Wk : Wv;
    (&Wl[0][0][0])[i] = (_Float16)src[j];
  }
  __syncthreads();

  const int w = tid >> 5, lane = tid & 31;
  const int half = lane >> 4, nn = lane & 15;
  const int row0 = blockIdx.x * 64 + w * 16;

#pragma unroll
  for (int mat = 0; mat < 3; ++mat) {
    const float* X = (mat == 0) ? Xq : (mat == 1) ? Xk : Xv;
    const float* bias = (mat == 0) ? bq : (mat == 1) ? bk : bv;
    _Float16* dst = (mat == 0) ? Q16 : (mat == 1) ? K16 : V16;

    // Build the two K=32 A-fragments directly from global f32 (cvt -> f16).
    // A 16x32 layout: lane m=L%16; elems 0..15 = k kc+half*16..kc+half*16+15
    const float* xr = X + (size_t)(row0 + nn) * 64;
    const v8f t0 = *(const v8f*)(xr + half * 16);
    const v8f t1 = *(const v8f*)(xr + 8 + half * 16);
    const v8f t2 = *(const v8f*)(xr + 32 + half * 16);
    const v8f t3 = *(const v8f*)(xr + 40 + half * 16);
    v16h a0, a1;
#pragma unroll
    for (int e = 0; e < 8; ++e) {
      a0[e] = (_Float16)t0[e]; a0[8 + e] = (_Float16)t1[e];
      a1[e] = (_Float16)t2[e]; a1[8 + e] = (_Float16)t3[e];
    }

#pragma unroll
    for (int nt = 0; nt < 4; ++nt) {
      const float bn = bias[nt * 16 + nn];
      v8f acc;
#pragma unroll
      for (int r = 0; r < 8; ++r) acc[r] = bn;  // bias depends on column only
      // B 32x16 layout: lane n=L%16; elems 0..15 = k half*16..half*16+15 of W row n.
      const _Float16* wrow = &Wl[mat][nt * 16 + nn][0];
      acc = wmma_f16(a0, *(const v16h*)(wrow + half * 16), acc);
      acc = wmma_f16(a1, *(const v16h*)(wrow + 32 + half * 16), acc);
#pragma unroll
      for (int r = 0; r < 8; ++r)  // C/D: lane half h holds rows h*8+r
        dst[(size_t)(row0 + half * 8 + r) * 64 + nt * 16 + nn] = (_Float16)acc[r];
    }
  }
}

// ---------------------------------------------------------------------------
// Kernel 2: causal flash attention, one (64-query block, head, batch) per WG.
// Wave = one 16-row M tile. head_dim 8 is zero-padded inside the K=32 WMMA.
// ---------------------------------------------------------------------------
__global__ void __launch_bounds__(128)
attn_flash(const _Float16* __restrict__ Q16, const _Float16* __restrict__ K16,
           const _Float16* __restrict__ V16, _Float16* __restrict__ O16) {
  __shared__ __align__(32) _Float16 Vt[16][64];      // V block transposed [dim][key]
  __shared__ __align__(32) _Float16 Pl[4][16][64];   // per-wave P tile staging

  const int tid = threadIdx.x, w = tid >> 5, lane = tid & 31;
  const int half = lane >> 4, nn = lane & 15;
  const int qb = blockIdx.x, h = blockIdx.y, bi = blockIdx.z;
  const int q0 = qb * 64;

  // Zero pad dims 8..15 of Vt once (B columns >= 8 are don't-care but finite).
  for (int i = tid; i < 8 * 64; i += 128) Vt[8 + (i >> 6)][i & 63] = (_Float16)0.f;

  // A fragment for Q (built once): only lanes half==0, elems 0..7 are live
  // (k = 0..7 == head_dim); everything else is the zero padding of K=32.
  v16h aq;
#pragma unroll
  for (int e = 0; e < 16; ++e) aq[e] = (_Float16)0.f;
  if (half == 0) {
    const v8h qv = *(const v8h*)(Q16 + (size_t)(bi * Lq + q0 + w * 16 + nn) * 64 + h * 8);
#pragma unroll
    for (int e = 0; e < 8; ++e) aq[e] = qv[e];
  }

  float m_r[8], l_r[8];
  v8f O;
#pragma unroll
  for (int r = 0; r < 8; ++r) { O[r] = 0.f; m_r[r] = -1e30f; l_r[r] = 0.f; }

  const float SC = 0.35355339059327373f;  // 1/sqrt(head_dim)
  const int waveRowMax = q0 + w * 16 + 15;
  const int nkb = qb + 1;                 // causal: key blocks 0..qb

  for (int kb = 0; kb < nkb; ++kb) {
    const int kbase = kb * 64;
    __syncthreads();  // protect previous Vt readers
    if (tid < 64) {   // stage V block transposed: Vt[d][key]
      const v8h vv = *(const v8h*)(V16 + (size_t)(bi * Sk + kbase + tid) * 64 + h * 8);
#pragma unroll
      for (int d = 0; d < 8; ++d) Vt[d][tid] = vv[d];
    }
    __syncthreads();

    if (kbase <= waveRowMax) {  // wave-uniform: EXEC stays all-1s for WMMA
      float sv[4][8];
#pragma unroll
      for (int nt = 0; nt < 4; ++nt) {
        v16h bk;
#pragma unroll
        for (int e = 0; e < 16; ++e) bk[e] = (_Float16)0.f;
        if (half == 0) {  // B[k][n] = K[kn][k]; row kn of K is contiguous in k
          const v8h kv = *(const v8h*)(K16 + (size_t)(bi * Sk + kbase + nt * 16 + nn) * 64 + h * 8);
#pragma unroll
          for (int e = 0; e < 8; ++e) bk[e] = kv[e];
        }
        v8f z;
#pragma unroll
        for (int r = 0; r < 8; ++r) z[r] = 0.f;
        const v8f s = wmma_f16(aq, bk, z);
        const int col = kbase + nt * 16 + nn;
#pragma unroll
        for (int r = 0; r < 8; ++r) {
          const int rowq = q0 + w * 16 + half * 8 + r;
          sv[nt][r] = (col <= rowq) ? s[r] * SC : -1e30f;  // causal mask
        }
      }

      // Online softmax: row stats live across the 16 lanes of each half.
      float bm[8];
#pragma unroll
      for (int r = 0; r < 8; ++r)
        bm[r] = fmaxf(fmaxf(sv[0][r], sv[1][r]), fmaxf(sv[2][r], sv[3][r]));
#pragma unroll
      for (int msk = 1; msk < 16; msk <<= 1)
#pragma unroll
        for (int r = 0; r < 8; ++r) bm[r] = fmaxf(bm[r], __shfl_xor(bm[r], msk));

      float al[8], rs[8];
#pragma unroll
      for (int r = 0; r < 8; ++r) {
        const float mn = fmaxf(m_r[r], bm[r]);
        al[r] = __expf(m_r[r] - mn);
        m_r[r] = mn;
        rs[r] = 0.f;
      }
#pragma unroll
      for (int nt = 0; nt < 4; ++nt)
#pragma unroll
        for (int r = 0; r < 8; ++r) {
          const float p = __expf(sv[nt][r] - m_r[r]);
          rs[r] += p;
          Pl[w][half * 8 + r][nt * 16 + nn] = (_Float16)p;  // C/D -> LDS
        }
#pragma unroll
      for (int msk = 1; msk < 16; msk <<= 1)
#pragma unroll
        for (int r = 0; r < 8; ++r) rs[r] += __shfl_xor(rs[r], msk);
#pragma unroll
      for (int r = 0; r < 8; ++r) {
        l_r[r] = l_r[r] * al[r] + rs[r];
        O[r] *= al[r];
      }

      // P (LDS) back as A-layout fragments; DS is in-order per wave, wait anyway.
      asm volatile("s_wait_dscnt 0x0" ::: "memory");
      const _Float16* prow = &Pl[w][nn][0];
#pragma unroll
      for (int c = 0; c < 2; ++c) {
        const v8h p0 = *(const v8h*)(prow + c * 32 + half * 16);
        const v8h p1 = *(const v8h*)(prow + c * 32 + 8 + half * 16);
        v16h ap;
#pragma unroll
        for (int e = 0; e < 8; ++e) { ap[e] = p0[e]; ap[8 + e] = p1[e]; }
        // B[k][n] = V[kbase+k][n] = Vt[n][k]: contiguous 32B per lane.
        const v16h bv = *(const v16h*)(&Vt[nn][0] + c * 32 + half * 16);
        O = wmma_f16(ap, bv, O);
      }
    }
  }

  if (nn < 8) {  // columns 8..15 of the O tile are padding
#pragma unroll
    for (int r = 0; r < 8; ++r) {
      const int rowq = q0 + w * 16 + half * 8 + r;
      O16[(size_t)(bi * Lq + rowq) * 64 + h * 8 + nn] = (_Float16)(O[r] / l_r[r]);
    }
  }
}

// ---------------------------------------------------------------------------
// Kernel 3: output projection, out = A @ Wo.T + bo, f32 to d_out.
// ---------------------------------------------------------------------------
__global__ void __launch_bounds__(128)
attn_out_proj(const _Float16* __restrict__ A16, const float* __restrict__ Wo,
              const float* __restrict__ bo, float* __restrict__ out) {
  __shared__ __align__(32) _Float16 Wl[64][64];
  const int tid = threadIdx.x;
  for (int i = tid; i < 64 * 64; i += 128) (&Wl[0][0])[i] = (_Float16)Wo[i];
  __syncthreads();

  const int w = tid >> 5, lane = tid & 31;
  const int half = lane >> 4, nn = lane & 15;
  const int row0 = blockIdx.x * 64 + w * 16;

  const _Float16* xr = A16 + (size_t)(row0 + nn) * 64;
  const v8h t0 = *(const v8h*)(xr + half * 16);
  const v8h t1 = *(const v8h*)(xr + 8 + half * 16);
  const v8h t2 = *(const v8h*)(xr + 32 + half * 16);
  const v8h t3 = *(const v8h*)(xr + 40 + half * 16);
  v16h a0, a1;
#pragma unroll
  for (int e = 0; e < 8; ++e) {
    a0[e] = t0[e]; a0[8 + e] = t1[e];
    a1[e] = t2[e]; a1[8 + e] = t3[e];
  }

#pragma unroll
  for (int nt = 0; nt < 4; ++nt) {
    const float bn = bo[nt * 16 + nn];
    v8f acc;
#pragma unroll
    for (int r = 0; r < 8; ++r) acc[r] = bn;
    const _Float16* wrow = &Wl[nt * 16 + nn][0];
    acc = wmma_f16(a0, *(const v16h*)(wrow + half * 16), acc);
    acc = wmma_f16(a1, *(const v16h*)(wrow + 32 + half * 16), acc);
#pragma unroll
    for (int r = 0; r < 8; ++r)
      out[(size_t)(row0 + half * 8 + r) * 64 + nt * 16 + nn] = acc[r];
  }
}

// ---------------------------------------------------------------------------
extern "C" void kernel_launch(void* const* d_in, const int* in_sizes, int n_in,
                              void* d_out, int out_size, void* d_ws, size_t ws_size,
                              hipStream_t stream) {
  (void)in_sizes; (void)n_in; (void)out_size; (void)ws_size;
  // setup_inputs order:
  // 0 queries, 1 keys, 2 values, 3 attention_mask,
  // 4 Wq, 5 bq, 6 Wk, 7 bk, 8 Wv, 9 bv, 10 Wo, 11 bo
  const float* queries = (const float*)d_in[0];
  const float* keys    = (const float*)d_in[1];
  const float* values  = (const float*)d_in[2];
  const float* Wq = (const float*)d_in[4];  const float* bq = (const float*)d_in[5];
  const float* Wk = (const float*)d_in[6];  const float* bk = (const float*)d_in[7];
  const float* Wv = (const float*)d_in[8];  const float* bv = (const float*)d_in[9];
  const float* Wo = (const float*)d_in[10]; const float* bo = (const float*)d_in[11];
  float* out = (float*)d_out;

  // Workspace: Q,K,V,AttnOut as f16 [b*L, 64] each = 4 * 1 MB = 4 MB.
  const size_t nElem = (size_t)Bb * Lq * Dm;
  _Float16* Q16 = (_Float16*)d_ws;
  _Float16* K16 = Q16 + nElem;
  _Float16* V16 = K16 + nElem;
  _Float16* A16 = V16 + nElem;

  const int rows = Bb * Lq;  // 8192
  attn_qkv_proj<<<dim3(rows / 64), dim3(128), 0, stream>>>(
      queries, keys, values, Wq, bq, Wk, bk, Wv, bv, Q16, K16, V16);
  attn_flash<<<dim3(Lq / 64, Hh, Bb), dim3(128), 0, stream>>>(Q16, K16, V16, A16);
  attn_out_proj<<<dim3(rows / 64), dim3(128), 0, stream>>>(A16, Wo, bo, out);
}
